// GCNLayer_22316650070499
// MI455X (gfx1250) — compile-verified, matrix operand order checked by reading, not run on hardware
//
#include <hip/hip_runtime.h>

// ---------------------------------------------------------------------------
// GCN layer for MI455X (gfx1250): out = (D^-1/2 (A+I) D^-1/2) X W^T + b
//   - pass 1: degrees -> dinv = rsqrt(rowsum+1)
//   - pass 2: prep_x: xs_t[b][f][m] = bf16(dinv[m]*x[b][m][f])  (LDS-tiled
//     transpose, coalesced both sides)
//   - pass 3: WMMA GEMM1 (adj @ xs) fused with WMMA GEMM2 (@ W^T) + bias;
//     X chunks staged into LDS with GLOBAL_LOAD_ASYNC_TO_LDS_B128 (ASYNCcnt),
//     double-buffered; A fragments streamed from global f32 adj and packed
//     with one v_perm_b32 per bf16 pair.
// ---------------------------------------------------------------------------

typedef __attribute__((ext_vector_type(16))) __bf16 v16bf;
typedef __attribute__((ext_vector_type(8)))  float  v8f;

#define FDIM 128   // feature dim (in == out)
#define MT   64    // output rows per block (4 waves x 16)
#define KC   64    // K chunk staged per iteration

__device__ __forceinline__ unsigned short bftrunc(float f) {
    return (unsigned short)(__builtin_bit_cast(unsigned, f) >> 16);
}

// D[15:0] = lo[31:16], D[31:16] = hi[31:16]  -> one v_perm_b32
__device__ __forceinline__ unsigned packtr(float lo, float hi) {
    return __builtin_amdgcn_perm(__builtin_bit_cast(unsigned, hi),
                                 __builtin_bit_cast(unsigned, lo),
                                 0x07060302u);
}

// Async global->LDS 16B copy (per lane), tracked by ASYNCcnt.
// LDS address for generic __shared__ pointers = low 32 bits (flat aperture).
__device__ __forceinline__ void async_copy_b128(void* lds_dst, const void* gsrc) {
    unsigned ldsa = (unsigned)(unsigned long long)lds_dst;
    asm volatile("global_load_async_to_lds_b128 %0, %1, off"
                 :: "v"(ldsa), "v"(gsrc) : "memory");
}

__device__ __forceinline__ void wait_async0() {
#if defined(__has_builtin)
#  if __has_builtin(__builtin_amdgcn_s_wait_asynccnt)
    __builtin_amdgcn_s_wait_asynccnt(0);
    return;
#  endif
#endif
    asm volatile("s_wait_asynccnt 0x0" ::: "memory");
}

union FragBF {
    v16bf v;
    uint4 q[2];
    unsigned int u[8];
};

// ---------------------------------------------------------------------------
// Pass 1: dinv[b,n] = rsqrt(sum_m adj[b,n,m] + 1)   (+1 = self loop)
// ---------------------------------------------------------------------------
__global__ void __launch_bounds__(256) degree_kernel(
    const float* __restrict__ adj, float* __restrict__ dinv, int N)
{
    int wave = threadIdx.x >> 5;
    int lane = threadIdx.x & 31;
    size_t row = (size_t)blockIdx.x * 8 + wave;
    const float* p = adj + row * (size_t)N;
    float s = 0.0f;
    for (int c = lane * 4; c < N; c += 128) {
        float4 v = *(const float4*)(p + c);
        s += v.x + v.y + v.z + v.w;
    }
    #pragma unroll
    for (int o = 16; o > 0; o >>= 1) s += __shfl_xor(s, o, 32);
    if (lane == 0) dinv[row] = rsqrtf(s + 1.0f);
}

// ---------------------------------------------------------------------------
// Pass 2: xs_t[b][f][m] = bf16(dinv[b][m] * x[b][m][f])
// 64x64 tiles through LDS; coalesced float4 reads, coalesced 8B writes.
// grid = (N/64, F/64, B), block = 256
// ---------------------------------------------------------------------------
__global__ void __launch_bounds__(256) prep_x_kernel(
    const float* __restrict__ x, const float* __restrict__ dinv,
    unsigned short* __restrict__ xs, int N)
{
    __shared__ unsigned short tile[64 * 72];   // padded rows (144B, 8B-aligned)
    int tid   = threadIdx.x;
    int mbase = blockIdx.x * 64;
    int fbase = blockIdx.y * 64;
    int bb    = blockIdx.z;
    const float* x_b    = x    + (size_t)bb * N * FDIM;
    const float* dinv_b = dinv + (size_t)bb * N;

    #pragma unroll
    for (int it = 0; it < 4; ++it) {
        int idx = tid + 256 * it;       // 1024 float4 in a 64x64 tile
        int m = idx >> 4;
        int c = (idx & 15) * 4;         // f_local
        float ds = dinv_b[mbase + m];
        float4 v = *(const float4*)(x_b + (size_t)(mbase + m) * FDIM + fbase + c);
        tile[(c + 0) * 72 + m] = bftrunc(v.x * ds);
        tile[(c + 1) * 72 + m] = bftrunc(v.y * ds);
        tile[(c + 2) * 72 + m] = bftrunc(v.z * ds);
        tile[(c + 3) * 72 + m] = bftrunc(v.w * ds);
    }
    __syncthreads();

    unsigned short* xs_b = xs + (size_t)bb * FDIM * N;
    #pragma unroll
    for (int it = 0; it < 4; ++it) {
        int idx = tid + 256 * it;       // 1024 uint2 (4 shorts each)
        int f = idx >> 4;
        int c = (idx & 15) * 4;         // m_local
        *(uint2*)(xs_b + (size_t)(fbase + f) * N + mbase + c) =
            *(const uint2*)(&tile[f * 72 + c]);
    }
}

// ---------------------------------------------------------------------------
// Pass 3: fused aggregate + linear.
// grid = (N/MT, B), block = 128 threads (4 waves); wave w owns rows
// [blockIdx.x*64 + 16w, +16) x all 128 features.
// ---------------------------------------------------------------------------
__global__ void __launch_bounds__(128) gcn_kernel(
    const float* __restrict__ x,             // [B,N,F] f32 (epilogue only)
    const float* __restrict__ adj,           // [B,N,N]
    const float* __restrict__ W,             // [F,F] row-major [o][f]
    const float* __restrict__ bias,          // [F]
    const float* __restrict__ dinv,          // [B,N]
    const unsigned short* __restrict__ xs,   // [B,F,N] bf16, pre-scaled
    float* __restrict__ out,                 // [B,N,F]
    int N)
{
    // Double-buffered X chunk [f][m] (bf16): B fragments are contiguous 16B
    // ds_load_b128 reads. Buffer 0 reused as Y staging after the K loop.
    __shared__ unsigned short lds_x[2 * FDIM * KC];  // 32 KB
    __shared__ unsigned short lds_w[FDIM * FDIM];    // 32 KB, bf16 [o][f]

    const int tid  = threadIdx.x;
    const int lane = tid & 31;
    const int wave = tid >> 5;
    const int lh   = (lane >> 4) & 1;   // lane half (0: lanes 0-15, 1: 16-31)
    const int ll   = lane & 15;
    const int bb   = blockIdx.y;
    const int rowbase = blockIdx.x * MT + wave * 16;
    const int irow = rowbase + ll;      // A-matrix row held by this lane

    const float* adj_b  = adj  + (size_t)bb * N * N;
    const float* x_b    = x    + (size_t)bb * N * FDIM;
    const float* dinv_b = dinv + (size_t)bb * N;
    const unsigned short* xs_b = xs + (size_t)bb * FDIM * N;

    // ---- stage W as bf16 row-major in LDS (once) ----
    #pragma unroll
    for (int it = 0; it < 32; ++it) {
        int idx = tid + 128 * it;         // 4096 float4 total
        int r = idx >> 5;                 // o
        int c = (idx & 31) * 4;           // f
        float4 v = *(const float4*)(W + (size_t)r * FDIM + c);
        *(uint2*)&lds_w[r * FDIM + c] =
            make_uint2(packtr(v.x, v.y), packtr(v.z, v.w));
    }

    // stage xs chunk -> LDS via async global->LDS b128 (no VGPR round-trip)
    auto stage_x = [&](int kb, int buf) {
        unsigned short* dst = &lds_x[buf * (FDIM * KC)];
        #pragma unroll
        for (int it = 0; it < 8; ++it) {
            int idx = tid + 128 * it;     // 1024 x 16B (16 KB)
            int f = idx >> 3;             // feature row
            int c = (idx & 7) * 8;        // m offset (8 shorts per 16B)
            async_copy_b128(dst + f * KC + c,
                            xs_b + (size_t)f * N + kb + c);
        }
    };

    v8f acc[8];
    #pragma unroll
    for (int t = 0; t < 8; ++t)
        #pragma unroll
        for (int r = 0; r < 8; ++r) acc[t][r] = 0.0f;

    // ============ GEMM 1: Y = adj @ (dinv*X), K over N, dbl-buffered =======
    stage_x(0, 0);
    wait_async0();
    __syncthreads();
    int cur = 0;
    for (int kb = 0; kb < N; kb += KC, cur ^= 1) {
        if (kb + KC < N) stage_x(kb + KC, cur ^ 1);   // prefetch next chunk

        const unsigned short* xbuf = &lds_x[cur * (FDIM * KC)];
        #pragma unroll
        for (int ks = 0; ks < KC / 32; ++ks) {
            // ---- A fragment straight from global adj ----
            // 16-bit A layout: lanes 0-15 hold K {0..7,16..23}, lanes 16-31
            // hold K {8..15,24..31}; lane's M = lane%16.
            int k0 = kb + ks * 32 + lh * 8;
            const float* ap = adj_b + (size_t)irow * N + k0;
            float4 a0 = *(const float4*)(ap + 0);
            float4 a1 = *(const float4*)(ap + 4);
            float4 a2 = *(const float4*)(ap + 16);
            float4 a3 = *(const float4*)(ap + 20);
            FragBF fa;
            fa.u[0] = packtr(a0.x, a0.y);
            fa.u[1] = packtr(a0.z, a0.w);
            fa.u[2] = packtr(a1.x, a1.y);
            fa.u[3] = packtr(a1.z, a1.w);
            fa.u[4] = packtr(a2.x, a2.y);
            fa.u[5] = packtr(a2.z, a2.w);
            fa.u[6] = packtr(a3.x, a3.y);
            fa.u[7] = packtr(a3.z, a3.w);

            // ---- B fragments from LDS; 8 feature tiles ----
            int kx0 = ks * 32 + lh * 8;
            #pragma unroll
            for (int t = 0; t < 8; ++t) {
                const unsigned short* bp = &xbuf[(t * 16 + ll) * KC + kx0];
                FragBF fb;
                fb.q[0] = *(const uint4*)(bp);
                fb.q[1] = *(const uint4*)(bp + 16);
                acc[t] = __builtin_amdgcn_wmma_f32_16x16x32_bf16(
                    false, fa.v, false, fb.v, (short)0, acc[t], false, false);
            }
        }
        wait_async0();     // our async stores to the other buffer are done
        __syncthreads();   // everyone staged & consumed -> flip buffers
    }

    // ---- epilogue: identity term + row scale, once per element ----
    // Y[i,f] = dinv[i] * ( sum_m adj[i,m]*dinv[m]*x[m,f] + dinv[i]*x[i,f] )
    float dsc[8];
    #pragma unroll
    for (int r = 0; r < 8; ++r) dsc[r] = dinv_b[rowbase + r + lh * 8];

    #pragma unroll
    for (int t = 0; t < 8; ++t)
        #pragma unroll
        for (int r = 0; r < 8; ++r) {
            int i = rowbase + r + lh * 8;          // C layout: M = r + 8*lh
            float xv = x_b[(size_t)i * FDIM + t * 16 + ll];
            acc[t][r] = (acc[t][r] + dsc[r] * xv) * dsc[r];
        }

    // ---- stage Y (bf16) into reused lds_x buffer 0 ----
    unsigned short* yw = &lds_x[wave * 16 * FDIM];   // per-wave 16x128 region
    #pragma unroll
    for (int t = 0; t < 8; ++t)
        #pragma unroll
        for (int r = 0; r < 8; ++r)
            yw[(r + lh * 8) * FDIM + t * 16 + ll] = bftrunc(acc[t][r]);
    __syncthreads();

    // ================= GEMM 2: Out = Y @ W^T + b  (K = 128) ================
    v8f acc2[8];
    #pragma unroll
    for (int t = 0; t < 8; ++t)
        #pragma unroll
        for (int r = 0; r < 8; ++r) acc2[t][r] = 0.0f;

    #pragma unroll
    for (int ks = 0; ks < FDIM / 32; ++ks) {
        int k0 = ks * 32 + lh * 8;
        const unsigned short* apy = &yw[ll * FDIM + k0];
        FragBF fa;
        fa.q[0] = *(const uint4*)(apy);
        fa.q[1] = *(const uint4*)(apy + 16);
        #pragma unroll
        for (int t = 0; t < 8; ++t) {
            // B[k=f][n=o] = W[o][f]: lane holds column o, contiguous f in LDS
            const unsigned short* bp = &lds_w[(t * 16 + ll) * FDIM + k0];
            FragBF fb;
            fb.q[0] = *(const uint4*)(bp);
            fb.q[1] = *(const uint4*)(bp + 16);
            acc2[t] = __builtin_amdgcn_wmma_f32_16x16x32_bf16(
                false, fa.v, false, fb.v, (short)0, acc2[t], false, false);
        }
    }

    // ---- bias + store ----
    float* ob = out + (size_t)bb * N * FDIM;
    #pragma unroll
    for (int t = 0; t < 8; ++t) {
        float bv = bias[t * 16 + ll];
        #pragma unroll
        for (int r = 0; r < 8; ++r) {
            int i = rowbase + r + lh * 8;
            ob[(size_t)i * FDIM + t * 16 + ll] = acc2[t][r] + bv;
        }
    }
}

// ---------------------------------------------------------------------------
extern "C" void kernel_launch(void* const* d_in, const int* in_sizes, int n_in,
                              void* d_out, int out_size, void* d_ws, size_t ws_size,
                              hipStream_t stream)
{
    const float* x    = (const float*)d_in[0];
    const float* adj  = (const float*)d_in[1];
    const float* W    = (const float*)d_in[2];
    const float* b    = (const float*)d_in[3];
    float* out = (float*)d_out;

    long long xsz = in_sizes[0];         // B*N*F
    long long asz = in_sizes[1];         // B*N*N
    int F = in_sizes[3];                 // 128
    int N = (int)((asz / xsz) * F);      // N = F * (N/F)
    int B = (int)(xsz / ((long long)N * F));

    // workspace: dinv (B*N f32) then xs_t (B*F*N bf16)
    float* dinv = (float*)d_ws;
    unsigned short* xs = (unsigned short*)((char*)d_ws +
                         (size_t)B * N * sizeof(float));

    degree_kernel<<<dim3((unsigned)((B * N) / 8)), dim3(256), 0, stream>>>(
        adj, dinv, N);
    prep_x_kernel<<<dim3((unsigned)(N / 64), (unsigned)(F / 64), (unsigned)B),
                    dim3(256), 0, stream>>>(x, dinv, xs, N);
    gcn_kernel<<<dim3((unsigned)(N / MT), (unsigned)B), dim3(128), 0, stream>>>(
        x, adj, W, b, dinv, xs, out, N);
}
